// ResidualMambaBlock_15178414424121
// MI455X (gfx1250) — matrix operand order repaired and checked by reading, not compiled, not run on hardware
//
#include <hip/hip_runtime.h>
#include <hip/hip_bf16.h>

// ---------------------------------------------------------------------------
// ResidualMambaBlock for MI455X (gfx1250, wave32, WMMA)
//   B=4, L=2048, D_MODEL=1024, D_INNER=2048, D_STATE=16, DT_RANK=64, D_CONV=4
// GEMMs: bf16 x bf16 -> f32 via v_wmma_f32_16x16x32_bf16, block tile 128x64,
// wave tile 32x32 (4 accumulators). Weights are pre-transposed to [N][K] so
// both LDS tiles and all fragment reads are contiguous 128-bit accesses.
// ---------------------------------------------------------------------------

#define BB   4
#define LL   2048
#define DM   1024
#define DI   2048
#define DS   16
#define DTR  64
#define ROWS (BB * LL)          // 8192

typedef __attribute__((ext_vector_type(16))) __bf16 v16bf;
typedef __attribute__((ext_vector_type(8)))  float  v8f;

union bfu    { unsigned short u; __bf16 b; };
union v16bfu { v16bf v; __bf16 e[16]; int4 q[2]; };
union v8fu   { v8f v; float e[8]; };

__device__ inline unsigned short f32_to_bf16_rne(float f) {
    unsigned int u = __float_as_uint(f);
    u += 0x7FFFu + ((u >> 16) & 1u);
    return (unsigned short)(u >> 16);
}
__device__ inline __bf16 f2bf(float f) { bfu t; t.u = f32_to_bf16_rne(f); return t.b; }

__device__ inline float sigmoidf_(float x) { return 1.0f / (1.0f + __expf(-x)); }
__device__ inline float siluf_(float x)    { return x * sigmoidf_(x); }
__device__ inline float gelu_tanh_(float x) {
    float x3 = x * x * x;
    return 0.5f * x * (1.0f + tanhf(0.7978845608f * (x + 0.044715f * x3)));
}
__device__ inline float softplusf_(float x) {
    return (x > 20.0f) ? x : log1pf(__expf(x));
}

// ------------------------------ elementwise --------------------------------

__global__ void cast_f32_bf16_kernel(const float* __restrict__ src,
                                     __bf16* __restrict__ dst, int n) {
    int i = blockIdx.x * blockDim.x + threadIdx.x;
    if (i < n) dst[i] = f2bf(src[i]);
}

// src[K][N] f32 -> dst[N][K] bf16  (weight transpose, one-time, tiny traffic)
__global__ void cast_transpose_bf16_kernel(const float* __restrict__ src,
                                           __bf16* __restrict__ dst,
                                           int K, int N) {
    int i = blockIdx.x * blockDim.x + threadIdx.x;
    if (i >= K * N) return;
    int k = i / N, n = i - k * N;
    dst[(size_t)n * K + k] = f2bf(src[i]);
}

// one block per row, D = 1024
__global__ void layernorm_bf16_kernel(const float* __restrict__ x,
                                      const float* __restrict__ g,
                                      const float* __restrict__ bta,
                                      __bf16* __restrict__ out, int D) {
    int row = blockIdx.x;
    const float* xr = x + (size_t)row * D;
    __shared__ float s1[256];
    __shared__ float s2[256];
    float a = 0.f, q = 0.f;
    for (int i = threadIdx.x; i < D; i += 256) {
        float v = xr[i];
        a += v; q += v * v;
    }
    s1[threadIdx.x] = a; s2[threadIdx.x] = q;
    __syncthreads();
    for (int off = 128; off > 0; off >>= 1) {
        if ((int)threadIdx.x < off) {
            s1[threadIdx.x] += s1[threadIdx.x + off];
            s2[threadIdx.x] += s2[threadIdx.x + off];
        }
        __syncthreads();
    }
    float mu  = s1[0] / (float)D;
    float var = s2[0] / (float)D - mu * mu;
    float inv = rsqrtf(var + 1e-5f);
    __bf16* orow = out + (size_t)row * D;
    for (int i = threadIdx.x; i < D; i += 256)
        orow[i] = f2bf((xr[i] - mu) * inv * g[i] + bta[i]);
}

// causal depthwise conv over time + bias + SiLU; reads xh-half of xz
__global__ void conv_silu_kernel(const float* __restrict__ xz,
                                 const float* __restrict__ conv_w,
                                 const float* __restrict__ conv_b,
                                 float* __restrict__ xh,
                                 __bf16* __restrict__ xh_bf) {
    int idx = blockIdx.x * blockDim.x + threadIdx.x;
    if (idx >= ROWS * DI) return;
    int c = idx & (DI - 1);
    int l = (idx >> 11) & (LL - 1);
    int b = idx / (DI * LL);
    const float* w = conv_w + c * 4;
    float s = conv_b[c];
#pragma unroll
    for (int k = 0; k < 4; k++) {
        int ls = l - 3 + k;
        if (ls >= 0)
            s += xz[((size_t)(b * LL + ls)) * (2 * DI) + c] * w[k];
    }
    float y = siluf_(s);
    xh[idx]    = y;
    xh_bf[idx] = f2bf(y);
}

// selective-state-space sequential scan; one thread per (batch, channel)
__global__ void ssm_scan_kernel(const float* __restrict__ dt,
                                const float* __restrict__ xh,
                                const float* __restrict__ xdbl,   // [rows][96]
                                const float* __restrict__ xz,     // z-half used
                                const float* __restrict__ A_log,
                                const float* __restrict__ Dp,
                                __bf16* __restrict__ ybf) {
    int t = blockIdx.x * blockDim.x + threadIdx.x;
    if (t >= BB * DI) return;
    int b = t >> 11;
    int c = t & (DI - 1);
    float Ar[DS], h[DS];
#pragma unroll
    for (int n = 0; n < DS; n++) {
        Ar[n] = -__expf(A_log[c * DS + n]);
        h[n]  = 0.f;
    }
    float Dc = Dp[c];
    for (int l = 0; l < LL; l++) {
        size_t r = (size_t)b * LL + l;
        float d  = dt[r * DI + c];
        float u  = xh[r * DI + c];
        const float* bc = xdbl + r * 96;
        float du = d * u;
        float y  = 0.f;
#pragma unroll
        for (int n = 0; n < DS; n++) {
            float dA = __expf(d * Ar[n]);
            h[n] = dA * h[n] + du * bc[64 + n];
            y   += h[n] * bc[80 + n];
        }
        y += Dc * u;
        float z = xz[r * (2 * DI) + DI + c];
        ybf[r * DI + c] = f2bf(y * siluf_(z));
    }
}

__global__ void residual1_kernel(const float* __restrict__ x,
                                 const float* __restrict__ delta,
                                 const float* __restrict__ mask,
                                 float* __restrict__ out, int n, int D) {
    int i = blockIdx.x * blockDim.x + threadIdx.x;
    if (i >= n) return;
    float m = mask[i / D];
    out[i] = x[i] + delta[i] * m;
}

__global__ void residual2_kernel(const float* __restrict__ x1,
                                 const float* __restrict__ z2,
                                 const float* __restrict__ mask,
                                 float* __restrict__ out, int n, int D) {
    int i = blockIdx.x * blockDim.x + threadIdx.x;
    if (i >= n) return;
    float m = mask[i / D];
    out[i] = (x1[i] + z2[i] * m) * m;
}

// ------------------------------ WMMA GEMM ----------------------------------
// C[M,N] f32 = A[M,K] bf16 * BT[N,K] bf16 (B pre-transposed) (+bias) (+act)
// 256 threads = 8 waves; block tile 128x64; wave tile 32x32 = 4 WMMA accs.
// ACT: 0=none, 1=gelu(tanh), 2=softplus. RAGGED: N not multiple of 64.
template <int ACT, bool HASB, bool RAGGED>
__global__ void gemm_bf16_wmma_kernel(const __bf16* __restrict__ A,
                                      const __bf16* __restrict__ BT,
                                      float* __restrict__ C,
                                      const float* __restrict__ bias,
                                      int M, int N, int K,
                                      int lda, int ldbt, int ldc) {
    __shared__ __align__(16) __bf16 Alds[128][40];  // 128 rows x 32 K (+pad)
    __shared__ __align__(16) __bf16 Btlds[64][40];  // 64 cols  x 32 K (+pad)

    const int m0   = blockIdx.y * 128;
    const int n0   = blockIdx.x * 64;
    const int tid  = threadIdx.x;
    const int lane = tid & 31;
    const int wv   = tid >> 5;
    const int wr   = wv & 3;   // 4 row-blocks of 32
    const int wc   = wv >> 2;  // 2 col-blocks of 32
    const int l15  = lane & 15;
    const int hi   = lane >> 4;

    v8fu acc00, acc01, acc10, acc11;
#pragma unroll
    for (int i = 0; i < 8; i++) {
        acc00.e[i] = 0.f; acc01.e[i] = 0.f;
        acc10.e[i] = 0.f; acc11.e[i] = 0.f;
    }

    // B-tile copy indices (64 rows of BT x 32 K = 256 int4)
    const int brow = tid >> 2, bseg = tid & 3;
    const bool brow_ok = RAGGED ? (n0 + brow < N) : true;

    for (int k0 = 0; k0 < K; k0 += 32) {
        __syncthreads();
        // A tile: 128x32 bf16 = 512 int4, 2 per thread (M mult of 128, K of 32)
#pragma unroll
        for (int it = 0; it < 2; it++) {
            int idx = tid + it * 256;
            int row = idx >> 2, seg = idx & 3;
            *(int4*)&Alds[row][seg * 8] =
                *(const int4*)(A + (size_t)(m0 + row) * lda + k0 + seg * 8);
        }
        // BT tile: 64x32 bf16 = 256 int4, 1 per thread; zero-fill ragged rows
        {
            int4 v = make_int4(0, 0, 0, 0);
            if (brow_ok)
                v = *(const int4*)(BT + (size_t)(n0 + brow) * ldbt + k0 + bseg * 8);
            *(int4*)&Btlds[brow][bseg * 8] = v;
        }
        if (k0 + 32 < K) {  // global_prefetch_b8 next K-slab (near-cache scope)
            __builtin_prefetch(A + (size_t)(m0 + (tid >> 2)) * lda + (k0 + 32), 0, 3);
            if (brow_ok)
                __builtin_prefetch(BT + (size_t)(n0 + brow) * ldbt + (k0 + 32), 0, 3);
        }
        __syncthreads();

        // A fragments (ISA 7.12.2 16-bit A): lane half hi holds K hi*8+0..7 in
        // VGPR0-3 and K 16+hi*8+0..7 in VGPR4-7 -> two contiguous 16B chunks.
        v16bfu a0, a1, b0, b1;
        const int ar0 = wr * 32 + l15;
        const int ar1 = ar0 + 16;
        const int akb = hi * 8;
        a0.q[0] = *(const int4*)&Alds[ar0][akb];
        a0.q[1] = *(const int4*)&Alds[ar0][akb + 16];
        a1.q[0] = *(const int4*)&Alds[ar1][akb];
        a1.q[1] = *(const int4*)&Alds[ar1][akb + 16];
        // B fragments: lane half hi holds K hi*8+0..7 and K 16+hi*8+0..7 for
        // its column (same K order as A) -> two contiguous 16B chunks.
        const int c0  = wc * 32 + l15;
        const int c1  = c0 + 16;
        const int bkb = hi * 8;
        b0.q[0] = *(const int4*)&Btlds[c0][bkb];
        b0.q[1] = *(const int4*)&Btlds[c0][bkb + 16];
        b1.q[0] = *(const int4*)&Btlds[c1][bkb];
        b1.q[1] = *(const int4*)&Btlds[c1][bkb + 16];

        acc00.v = __builtin_amdgcn_wmma_f32_16x16x32_bf16(
            false, a0.v, false, b0.v, (short)0, acc00.v, false, false);
        acc01.v = __builtin_amdgcn_wmma_f32_16x16x32_bf16(
            false, a0.v, false, b1.v, (short)0, acc01.v, false, false);
        acc10.v = __builtin_amdgcn_wmma_f32_16x16x32_bf16(
            false, a1.v, false, b0.v, (short)0, acc10.v, false, false);
        acc11.v = __builtin_amdgcn_wmma_f32_16x16x32_bf16(
            false, a1.v, false, b1.v, (short)0, acc11.v, false, false);
    }

    // store C (16x16 f32 acc layout: lanes 0-15 rows 0-7, lanes 16-31 rows 8-15)
    v8fu* accs[2][2] = { { &acc00, &acc01 }, { &acc10, &acc11 } };
#pragma unroll
    for (int rb = 0; rb < 2; rb++) {
#pragma unroll
        for (int cb = 0; cb < 2; cb++) {
            int col = n0 + wc * 32 + cb * 16 + l15;
            if (!RAGGED || col < N) {
                float bia = HASB ? bias[col] : 0.f;
                v8fu* acc = accs[rb][cb];
#pragma unroll
                for (int i = 0; i < 8; i++) {
                    int row = m0 + wr * 32 + rb * 16 + hi * 8 + i;
                    float v = acc->e[i] + bia;
                    if (ACT == 1)      v = gelu_tanh_(v);
                    else if (ACT == 2) v = softplusf_(v);
                    C[(size_t)row * ldc + col] = v;
                }
            }
        }
    }
}

// ------------------------------ launcher -----------------------------------

extern "C" void kernel_launch(void* const* d_in, const int* in_sizes, int n_in,
                              void* d_out, int out_size, void* d_ws, size_t ws_size,
                              hipStream_t stream) {
    const float* x      = (const float*)d_in[0];
    const float* mask   = (const float*)d_in[1];
    const float* g1     = (const float*)d_in[2];
    const float* b1     = (const float*)d_in[3];
    const float* W_in   = (const float*)d_in[4];
    const float* conv_w = (const float*)d_in[5];
    const float* conv_b = (const float*)d_in[6];
    const float* W_x    = (const float*)d_in[7];
    const float* W_dt   = (const float*)d_in[8];
    const float* b_dt   = (const float*)d_in[9];
    const float* A_log  = (const float*)d_in[10];
    const float* Dp     = (const float*)d_in[11];
    const float* W_out  = (const float*)d_in[12];
    const float* g2     = (const float*)d_in[13];
    const float* b2     = (const float*)d_in[14];
    const float* W1     = (const float*)d_in[15];
    const float* bf1    = (const float*)d_in[16];
    const float* W2     = (const float*)d_in[17];
    const float* bf2    = (const float*)d_in[18];
    float* out = (float*)d_out;

    size_t off = 0;
    auto alloc = [&](size_t bytes) -> void* {
        void* p = (char*)d_ws + off;
        off += (bytes + 255) & ~(size_t)255;
        return p;
    };

    __bf16* XN_BF   = (__bf16*)alloc((size_t)ROWS * DM * 2);
    __bf16* WIN_BT  = (__bf16*)alloc((size_t)DM * 2 * DI * 2);   // [4096][1024]
    __bf16* WX_BT   = (__bf16*)alloc((size_t)DI * 96 * 2);       // [96][2048]
    __bf16* WDT_BT  = (__bf16*)alloc((size_t)DTR * DI * 2);      // [2048][64]
    __bf16* WOUT_BT = (__bf16*)alloc((size_t)DI * DM * 2);       // [1024][2048]
    __bf16* W1_BT   = (__bf16*)alloc((size_t)DM * 4 * DM * 2);   // [4096][1024]
    __bf16* W2_BT   = (__bf16*)alloc((size_t)4 * DM * DM * 2);   // [1024][4096]
    float*  XZ      = (float*) alloc((size_t)ROWS * 2 * DI * 4); // reused: HID
    float*  XH      = (float*) alloc((size_t)ROWS * DI * 4);
    __bf16* XH_BF   = (__bf16*)alloc((size_t)ROWS * DI * 2);
    float*  XDBL    = (float*) alloc((size_t)ROWS * 96 * 4);
    __bf16* XDBL_BF = (__bf16*)alloc((size_t)ROWS * 96 * 2);
    float*  DT      = (float*) alloc((size_t)ROWS * DI * 4);     // reused: HID_BF
    __bf16* Y_BF    = (__bf16*)alloc((size_t)ROWS * DI * 2);
    float*  MO      = (float*) alloc((size_t)ROWS * DM * 4);     // reused: Z2
    float*  X1      = (float*) alloc((size_t)ROWS * DM * 4);
    __bf16* X1N_BF  = (__bf16*)alloc((size_t)ROWS * DM * 2);
    float*  HID     = XZ;                 // [ROWS, 4096] f32
    __bf16* HID_BF  = (__bf16*)DT;        // [ROWS, 4096] bf16
    float*  Z2      = MO;

    auto cast = [&](const float* s, __bf16* d, int n) {
        cast_f32_bf16_kernel<<<(n + 255) / 256, 256, 0, stream>>>(s, d, n);
    };
    auto castT = [&](const float* s, __bf16* d, int K, int N) {
        int n = K * N;
        cast_transpose_bf16_kernel<<<(n + 255) / 256, 256, 0, stream>>>(s, d, K, N);
    };
    auto gemm = [&](const __bf16* A, const __bf16* Bt, float* Cm, const float* bias,
                    int M, int N, int K, int lda, int ldbt, int ldc, int act) {
        dim3 grid((N + 63) / 64, (M + 127) / 128);
        bool ragged = (N % 64) != 0;
        if (ragged) {
            // only the no-bias, no-act GEMM (W_x) is ragged in this model
            gemm_bf16_wmma_kernel<0, false, true><<<grid, 256, 0, stream>>>(
                A, Bt, Cm, nullptr, M, N, K, lda, ldbt, ldc);
        } else if (bias) {
            if (act == 1)
                gemm_bf16_wmma_kernel<1, true, false><<<grid, 256, 0, stream>>>(
                    A, Bt, Cm, bias, M, N, K, lda, ldbt, ldc);
            else if (act == 2)
                gemm_bf16_wmma_kernel<2, true, false><<<grid, 256, 0, stream>>>(
                    A, Bt, Cm, bias, M, N, K, lda, ldbt, ldc);
            else
                gemm_bf16_wmma_kernel<0, true, false><<<grid, 256, 0, stream>>>(
                    A, Bt, Cm, bias, M, N, K, lda, ldbt, ldc);
        } else {
            gemm_bf16_wmma_kernel<0, false, false><<<grid, 256, 0, stream>>>(
                A, Bt, Cm, nullptr, M, N, K, lda, ldbt, ldc);
        }
    };

    // weight cast+transpose (one-time; ~30 MB total vs ~1 GB activations)
    castT(W_in,  WIN_BT,  DM, 2 * DI);
    castT(W_x,   WX_BT,   DI, 96);
    castT(W_dt,  WDT_BT,  DTR, DI);
    castT(W_out, WOUT_BT, DI, DM);
    castT(W1,    W1_BT,   DM, 4 * DM);
    castT(W2,    W2_BT,   4 * DM, DM);

    // norm1 -> bf16
    layernorm_bf16_kernel<<<ROWS, 256, 0, stream>>>(x, g1, b1, XN_BF, DM);

    // xz = xn @ W_in : [8192,4096]
    gemm(XN_BF, WIN_BT, XZ, nullptr, ROWS, 2 * DI, DM, DM, DM, 2 * DI, 0);

    // causal depthwise conv + SiLU
    conv_silu_kernel<<<(ROWS * DI + 255) / 256, 256, 0, stream>>>(
        XZ, conv_w, conv_b, XH, XH_BF);

    // x_dbl = xh @ W_x : [8192,96]
    gemm(XH_BF, WX_BT, XDBL, nullptr, ROWS, 96, DI, DI, DI, 96, 0);
    cast(XDBL, XDBL_BF, ROWS * 96);

    // dt = softplus(dt_lo @ W_dt + b_dt) : [8192,2048]  (A lda=96, 64 cols)
    gemm(XDBL_BF, WDT_BT, DT, b_dt, ROWS, DI, DTR, 96, DTR, DI, 2);

    // sequential scan -> y * silu(z) as bf16
    ssm_scan_kernel<<<(BB * DI + 255) / 256, 256, 0, stream>>>(
        DT, XH, XDBL, XZ, A_log, Dp, Y_BF);

    // mamba out = y @ W_out : [8192,1024]
    gemm(Y_BF, WOUT_BT, MO, nullptr, ROWS, DM, DI, DI, DI, DM, 0);

    // x1 = x + mo * mask
    residual1_kernel<<<(ROWS * DM + 255) / 256, 256, 0, stream>>>(
        x, MO, mask, X1, ROWS * DM, DM);

    // norm2 -> bf16
    layernorm_bf16_kernel<<<ROWS, 256, 0, stream>>>(X1, g2, b2, X1N_BF, DM);

    // hid = gelu(x1n @ W1 + bf1) : [8192,4096]
    gemm(X1N_BF, W1_BT, HID, bf1, ROWS, 4 * DM, DM, DM, DM, 4 * DM, 1);
    cast(HID, HID_BF, ROWS * 4 * DM);

    // z2 = hid @ W2 + bf2 : [8192,1024]
    gemm(HID_BF, W2_BT, Z2, bf2, ROWS, DM, 4 * DM, 4 * DM, 4 * DM, DM, 0);

    // out = (x1 + z2*mask) * mask
    residual2_kernel<<<(ROWS * DM + 255) / 256, 256, 0, stream>>>(
        X1, Z2, mask, out, ROWS * DM, DM);

    (void)in_sizes; (void)n_in; (void)out_size; (void)ws_size;
}
